// Stage1_12197707120912
// MI455X (gfx1250) — matrix operand order repaired and checked by reading, not compiled, not run on hardware
//
#include <hip/hip_runtime.h>

#define BB 16
#define SS 2048
#define DD 128
#define HH 4
#define HDD 32
#define MM (BB * SS) // 32768 rows

typedef __attribute__((ext_vector_type(16))) _Float16 v16h;
typedef __attribute__((ext_vector_type(8)))  _Float16 v8h;
typedef __attribute__((ext_vector_type(8)))  float    v8f;

// ---------------- WMMA fragment helpers (ISA 7.12.2 layouts) ----------------

// A-matrix 16x32 f16 (MxK): lane half hf: elem j<8 -> K = hf*8+j ; j>=8 -> K = 16+hf*8+(j-8)
__device__ __forceinline__ v16h load_a(const _Float16* base, int m0, int k0, int ld) {
  const int lane = threadIdx.x & 31;
  const int hf = lane >> 4, r = lane & 15;
  const _Float16* p = base + (size_t)(m0 + r) * ld + k0 + hf * 8;
  v8h lo = *(const v8h*)p;
  v8h hi = *(const v8h*)(p + 16);
  return __builtin_shufflevector(lo, hi, 0,1,2,3,4,5,6,7,8,9,10,11,12,13,14,15);
}

// B-matrix 32x16 f16 (KxN): lane%16 -> column, lane half hf: elem j<8 -> K = hf*8+j ; j>=8 -> K = 16+hf*8+(j-8)
// Storage must be K-contiguous per column (transposed weights / row-major K,Q / Vt).
__device__ __forceinline__ v16h load_b(const _Float16* base, int n0, int k0, int ld) {
  const int lane = threadIdx.x & 31;
  const int hf = lane >> 4, c = lane & 15;
  const _Float16* p = base + (size_t)(n0 + c) * ld + k0 + hf * 8;
  v8h lo = *(const v8h*)p;
  v8h hi = *(const v8h*)(p + 16);
  return __builtin_shufflevector(lo, hi, 0,1,2,3,4,5,6,7,8,9,10,11,12,13,14,15);
}

__device__ __forceinline__ v8f wmma16(v16h a, v16h b, v8f c) {
  return __builtin_amdgcn_wmma_f32_16x16x32_f16(false, a, false, b, (short)0, c, false, false);
}

__device__ __forceinline__ v8f vzero8() {
  v8f z;
#pragma unroll
  for (int i = 0; i < 8; ++i) z[i] = 0.f;
  return z;
}

// ---------------- conversion kernels ----------------

__global__ __launch_bounds__(256) void k_f32_to_f16(const float* __restrict__ s,
                                                    _Float16* __restrict__ d, int n) {
  int i = blockIdx.x * 256 + threadIdx.x;
  if (i < n) d[i] = (_Float16)s[i];
}

// Wt[c*128 + k] = (f16) W[k*128 + c]   (K-contiguous per output column)
__global__ __launch_bounds__(256) void k_wT_f16(const float* __restrict__ W,
                                                _Float16* __restrict__ Wt) {
  int i = blockIdx.x * 256 + threadIdx.x; // 0..16383
  int k = i >> 7, c = i & 127;
  Wt[(size_t)c * DD + k] = (_Float16)W[(size_t)k * DD + c];
}

// ---------------- generic 128-wide WMMA GEMM ----------------
// Block: 128 rows x 128 cols, 8 waves; wave = 32 rows x 64 cols (8 WMMA per k-step).
// MODE 0: f16 row-major   MODE 1: f16 V-transposed [b,h,hd,s]
// MODE 2: f32 row-major   MODE 3: relu -> f16 row-major
template <int MODE>
__global__ __launch_bounds__(256) void k_gemm(const _Float16* __restrict__ A,
                                              const _Float16* __restrict__ Wt,
                                              const float* __restrict__ bias,
                                              _Float16* __restrict__ outh,
                                              float* __restrict__ outf) {
  const int wave = threadIdx.x >> 5;
  const int lane = threadIdx.x & 31;
  const int r0 = blockIdx.x * 128 + (wave & 3) * 32;
  const int c0 = (wave >> 2) * 64;

  v8f acc[2][4];
#pragma unroll
  for (int rt = 0; rt < 2; ++rt)
#pragma unroll
    for (int nt = 0; nt < 4; ++nt) acc[rt][nt] = vzero8();

#pragma unroll
  for (int ks = 0; ks < DD; ks += 32) {
    v16h a0 = load_a(A, r0, ks, DD);
    v16h a1 = load_a(A, r0 + 16, ks, DD);
#pragma unroll
    for (int nt = 0; nt < 4; ++nt) {
      v16h b = load_b(Wt, c0 + nt * 16, ks, DD);
      acc[0][nt] = wmma16(a0, b, acc[0][nt]);
      acc[1][nt] = wmma16(a1, b, acc[1][nt]);
    }
  }

  const int cb = lane & 15;
  const int rb = (lane >> 4) * 8;
#pragma unroll
  for (int rt = 0; rt < 2; ++rt) {
#pragma unroll
    for (int nt = 0; nt < 4; ++nt) {
      const int col = c0 + nt * 16 + cb;
      const float bv = bias[col];
#pragma unroll
      for (int i = 0; i < 8; ++i) {
        const int row = r0 + rt * 16 + rb + i;
        float v = acc[rt][nt][i] + bv;
        if (MODE == 0) {
          outh[(size_t)row * DD + col] = (_Float16)v;
        } else if (MODE == 1) {
          const int h = col >> 5, hd = col & 31;
          const int b = row >> 11, s = row & (SS - 1);
          outh[((size_t)((b * HH + h) * HDD + hd)) * SS + s] = (_Float16)v;
        } else if (MODE == 2) {
          outf[(size_t)row * DD + col] = v;
        } else {
          outh[(size_t)row * DD + col] = (_Float16)(v > 0.f ? v : 0.f);
        }
      }
    }
  }
}

// ---------------- flash attention, transposed-score formulation ----------------
// S^T = K_tile(A) x Q^T(B): softmax axis (kv) lands in VGPR elems + lane halves,
// so max/sum reductions need ONE xor-16 shuffle each. The S^T C-fragment is
// bit-identical to the A-fragment of O = P x V (K=32), so P feeds the second
// WMMA with zero cross-lane traffic (just v_cvt_pk f32->f16 packing).
__global__ __launch_bounds__(128) void k_attn(const _Float16* __restrict__ q,
                                              const _Float16* __restrict__ k,
                                              const _Float16* __restrict__ vt,
                                              _Float16* __restrict__ attn) {
  __shared__ float lsh[4][16];
  const int wave = threadIdx.x >> 5;
  const int lane = threadIdx.x & 31;
  const int b = blockIdx.z, h = blockIdx.y;
  const int q0 = blockIdx.x * 64 + wave * 16;
  const int rowbase = b * SS;
  const int hvbase = (b * HH + h) * HDD;

  // Q fragment, pre-scaled by (1/sqrt(HD)) * log2(e) so scores are in exp2 domain
  v16h bq = load_b(q, rowbase + q0, h * HDD, DD);
  const _Float16 sc = (_Float16)(0.17677669529663687f * 1.4426950408889634f);
#pragma unroll
  for (int i = 0; i < 16; ++i) bq[i] *= sc;

  float m = -1e30f, l = 0.f;
  v8f o0 = vzero8(), o1 = vzero8();

  for (int kv = 0; kv < SS; kv += 64) {
    // prefetch V B-fragments for both 32-kv chunks (consumed after the softmax VALU)
    v16h bv[2][2];
#pragma unroll
    for (int c = 0; c < 2; ++c) {
      bv[c][0] = load_b(vt, hvbase + 0,  kv + 32 * c, SS);
      bv[c][1] = load_b(vt, hvbase + 16, kv + 32 * c, SS);
    }

    // 4 transposed score tiles: rows kv+16t..kv+16t+15, cols = q
    v8f st[4];
#pragma unroll
    for (int t = 0; t < 4; ++t) {
      v16h ak = load_a(k, rowbase + kv + 16 * t, h * HDD, DD);
      st[t] = wmma16(ak, bq, vzero8());
    }

    // tree column-max over 64 kv (depth 5) + one xor16
    float e[8];
#pragma unroll
    for (int i = 0; i < 8; ++i)
      e[i] = fmaxf(fmaxf(st[0][i], st[1][i]), fmaxf(st[2][i], st[3][i]));
#pragma unroll
    for (int i = 0; i < 4; ++i) e[i] = fmaxf(e[i], e[i + 4]);
    float mt = fmaxf(fmaxf(e[0], e[2]), fmaxf(e[1], e[3]));
    mt = fmaxf(mt, __shfl_xor(mt, 16, 32));

    const float mn = fmaxf(m, mt);
    const float alpha = __builtin_amdgcn_exp2f(m - mn);
    m = mn;

#pragma unroll
    for (int t = 0; t < 4; ++t)
#pragma unroll
      for (int i = 0; i < 8; ++i)
        st[t][i] = __builtin_amdgcn_exp2f(st[t][i] - mn);

    // tree column-sum + one xor16
    float r[8];
#pragma unroll
    for (int i = 0; i < 8; ++i)
      r[i] = (st[0][i] + st[1][i]) + (st[2][i] + st[3][i]);
#pragma unroll
    for (int i = 0; i < 4; ++i) r[i] += r[i + 4];
    float rs = (r[0] + r[2]) + (r[1] + r[3]);
    rs += __shfl_xor(rs, 16, 32);
    l = l * alpha + rs;
#pragma unroll
    for (int i = 0; i < 8; ++i) { o0[i] *= alpha; o1[i] *= alpha; }

    // P tiles: pack scores straight into A-fragments (C-layout == A-layout)
#pragma unroll
    for (int c = 0; c < 2; ++c) {
      union { int i[8]; v16h v; } ap;
#pragma unroll
      for (int d = 0; d < 4; ++d) {
        ap.i[d]     = __builtin_bit_cast(int,
            __builtin_amdgcn_cvt_pkrtz(st[2 * c][2 * d], st[2 * c][2 * d + 1]));
        ap.i[4 + d] = __builtin_bit_cast(int,
            __builtin_amdgcn_cvt_pkrtz(st[2 * c + 1][2 * d], st[2 * c + 1][2 * d + 1]));
      }
      o0 = wmma16(ap.v, bv[c][0], o0);
      o1 = wmma16(ap.v, bv[c][1], o1);
    }
  }

  // l lives at lane%16 == q; O rows need it per element -> hop via 16-float LDS slot
  if (lane < 16) lsh[wave][lane] = l;
  asm volatile("s_wait_dscnt 0" ::: "memory"); // same-wave LDS RAW (in-order DS pipe)
  const int cb = lane & 15;
  const int rb = (lane >> 4) * 8;
  float linv[8];
#pragma unroll
  for (int i = 0; i < 8; ++i) linv[i] = 1.f / lsh[wave][rb + i];

  _Float16* op = attn + (size_t)(rowbase + q0 + rb) * DD + h * HDD + cb;
#pragma unroll
  for (int i = 0; i < 8; ++i) {
    op[(size_t)i * DD]      = (_Float16)(o0[i] * linv[i]);
    op[(size_t)i * DD + 16] = (_Float16)(o1[i] * linv[i]);
  }
}

// ---------------- layernorm(a + res) * g + be ; optional f32/f16 outputs ----------------
__global__ __launch_bounds__(256) void k_ln(const float* __restrict__ a,
                                            const float* __restrict__ res,
                                            const float* __restrict__ g,
                                            const float* __restrict__ be,
                                            float* __restrict__ outf,
                                            _Float16* __restrict__ outh) {
  const int wave = threadIdx.x >> 5, lane = threadIdx.x & 31;
  const size_t row = (size_t)blockIdx.x * 8 + wave;
  const float4 xa = *(const float4*)(a + row * DD + lane * 4);
  const float4 xr = *(const float4*)(res + row * DD + lane * 4);
  float x[4] = {xa.x + xr.x, xa.y + xr.y, xa.z + xr.z, xa.w + xr.w};
  float s = x[0] + x[1] + x[2] + x[3];
  float s2 = x[0]*x[0] + x[1]*x[1] + x[2]*x[2] + x[3]*x[3];
#pragma unroll
  for (int d = 1; d < 32; d <<= 1) {
    s  += __shfl_xor(s, d, 32);
    s2 += __shfl_xor(s2, d, 32);
  }
  const float mean = s * (1.f / 128.f);
  const float var = s2 * (1.f / 128.f) - mean * mean;
  const float rstd = rsqrtf(var + 1e-5f);
#pragma unroll
  for (int j = 0; j < 4; ++j) {
    const int col = lane * 4 + j;
    const float y = (x[j] - mean) * rstd * g[col] + be[col];
    if (outf) outf[row * DD + col] = y;
    if (outh) outh[row * DD + col] = (_Float16)y;
  }
}

// ---------------- host orchestration ----------------
extern "C" void kernel_launch(void* const* d_in, const int* in_sizes, int n_in,
                              void* d_out, int out_size, void* d_ws, size_t ws_size,
                              hipStream_t stream) {
  const float* x   = (const float*)d_in[0];
  const float* Wq  = (const float*)d_in[1];
  const float* bq  = (const float*)d_in[2];
  const float* Wk  = (const float*)d_in[3];
  const float* bk  = (const float*)d_in[4];
  const float* Wv  = (const float*)d_in[5];
  const float* bv  = (const float*)d_in[6];
  const float* Wo  = (const float*)d_in[7];
  const float* bo  = (const float*)d_in[8];
  const float* g1  = (const float*)d_in[9];
  const float* b1  = (const float*)d_in[10];
  const float* W1  = (const float*)d_in[11];
  const float* bf1 = (const float*)d_in[12];
  const float* W2  = (const float*)d_in[13];
  const float* bf2 = (const float*)d_in[14];
  const float* g2  = (const float*)d_in[15];
  const float* b2  = (const float*)d_in[16];
  float* out = (float*)d_out;

  char* ws = (char*)d_ws;
  const size_t R = (size_t)MM * DD * sizeof(_Float16); // 8 MB region
  _Float16* xh   = (_Float16*)(ws);           // later: h1 (relu output)
  _Float16* qh   = (_Float16*)(ws + R);       // later: o/ff2 f32 (spans 2 regions)
  _Float16* kh   = (_Float16*)(ws + 2 * R);
  _Float16* vth  = (_Float16*)(ws + 3 * R);   // later: attended f16
  _Float16* ath  = (_Float16*)(ws + 4 * R);   // attn output f16
  float*    of   = (float*)(ws + R);          // 16 MB f32 (aliases qh+kh); reused for ff2
  _Float16* wtb  = (_Float16*)(ws + 5 * R);   // 6 x 128x128 f16 transposed weights
  _Float16* WqT = wtb + 0 * DD * DD;
  _Float16* WkT = wtb + 1 * DD * DD;
  _Float16* WvT = wtb + 2 * DD * DD;
  _Float16* WoT = wtb + 3 * DD * DD;
  _Float16* W1T = wtb + 4 * DD * DD;
  _Float16* W2T = wtb + 5 * DD * DD;

  // 1) convert weights (transposed) + activations to f16
  k_wT_f16<<<64, 256, 0, stream>>>(Wq, WqT);
  k_wT_f16<<<64, 256, 0, stream>>>(Wk, WkT);
  k_wT_f16<<<64, 256, 0, stream>>>(Wv, WvT);
  k_wT_f16<<<64, 256, 0, stream>>>(Wo, WoT);
  k_wT_f16<<<64, 256, 0, stream>>>(W1, W1T);
  k_wT_f16<<<64, 256, 0, stream>>>(W2, W2T);
  k_f32_to_f16<<<(MM * DD) / 256, 256, 0, stream>>>(x, xh, MM * DD);

  // 2) QKV projections
  k_gemm<0><<<MM / 128, 256, 0, stream>>>(xh, WqT, bq, qh, nullptr);
  k_gemm<0><<<MM / 128, 256, 0, stream>>>(xh, WkT, bk, kh, nullptr);
  k_gemm<1><<<MM / 128, 256, 0, stream>>>(xh, WvT, bv, vth, nullptr);

  // 3) flash attention
  dim3 ag(SS / 64, HH, BB);
  k_attn<<<ag, 128, 0, stream>>>(qh, kh, vth, ath);

  // 4) output projection (f32) + residual LN -> attended (f16, reuses vt region)
  k_gemm<2><<<MM / 128, 256, 0, stream>>>(ath, WoT, bo, nullptr, of);
  k_ln<<<MM / 8, 256, 0, stream>>>(of, x, g1, b1, nullptr, vth);

  // 5) FFN: relu(attended@W1+bf1) -> h1 (reuses xh); h1@W2+bf2 -> ff2 (reuses of)
  k_gemm<3><<<MM / 128, 256, 0, stream>>>(vth, W1T, bf1, xh, nullptr);
  k_gemm<2><<<MM / 128, 256, 0, stream>>>(xh, W2T, bf2, nullptr, of);

  // 6) final LN(ff2 + x) -> d_out (f32)
  k_ln<<<MM / 8, 256, 0, stream>>>(of, x, g2, b2, out, nullptr);
}
